// ScagRcsmFusion_88364657148135
// MI455X (gfx1250) — matrix operand order repaired and checked by reading, not compiled, not run on hardware
//
#include <hip/hip_runtime.h>
#include <math.h>

// ---------------------------------------------------------------------------
// ScagRcsmFusion for MI455X (gfx1250): all large GEMMs via bf16 WMMA
// (v_wmma_f32_16x16x32_bf16, fp32 accumulate), guard-free b128 fragment loads,
// 32x32 tile per wave (4 wmma / K-step). Everything else VALU kernels.
// ---------------------------------------------------------------------------

typedef __attribute__((ext_vector_type(16))) __bf16 v16bf;
typedef __attribute__((ext_vector_type(8)))  float  v8f;
typedef unsigned short u16;
typedef unsigned int   u32;
typedef __attribute__((ext_vector_type(4)))  u32    v4u;

#define NB    1024          // batch
#define DIMX  768
#define DIX   1536          // 2*DIM
#define LX    4
#define RX    (NB * LX)     // 4096 rows
#define PJLD  96            // proj row stride (80 padded to 96)
#define CDIV(a, b) (((a) + (b) - 1) / (b))

// ---------------- bf16 helpers -------------------------------------------

__device__ __forceinline__ u16 f2bf(float f) {
  u32 u = __float_as_uint(f);
  u32 lsb = (u >> 16) & 1u;
  u += 0x7fffu + lsb;               // round-to-nearest-even
  return (u16)(u >> 16);
}

union FragAB { v16bf v; u32 u[8]; v4u q[2]; };

// A fragment: 16x32 bf16, row-major A[m][k], lda elements (multiple of 8).
// Lane l: m = m0 + (l&15); K-base kb = (l&16)?8:0.
// halves: dwords 0..3 = K kb+0..7 (contiguous), dwords 4..7 = K kb+16..23.
__device__ __forceinline__ v16bf load_a_frag(const u16* __restrict__ A, int lda,
                                             int m0, int k0, int lane) {
  int m  = m0 + (lane & 15);
  int kb = (lane & 16) ? 8 : 0;
  const u16* p = A + (size_t)m * lda + k0 + kb;
  FragAB f;
  f.q[0] = *(const v4u*)p;          // 8 halves: K kb+0..7
  f.q[1] = *(const v4u*)(p + 16);   // 8 halves: K kb+16..23
  return f.v;
}

// B fragment from pre-transposed weight Wt[n][k] (N x K row-major, K mult of 8).
// Lane l: n = n0 + (l&15); K-base kg = (l&16)?8:0; halves K kg+0..7, kg+16..23.
__device__ __forceinline__ v16bf load_b_frag(const u16* __restrict__ Wt, int K,
                                             int n0, int k0, int lane) {
  int kg = (lane & 16) ? 8 : 0;
  const u16* p = Wt + (size_t)(n0 + (lane & 15)) * K + k0 + kg;
  FragAB f;
  f.q[0] = *(const v4u*)p;
  f.q[1] = *(const v4u*)(p + 16);
  return f.v;
}

__device__ __forceinline__ float act_softplus(float x) {
  return (x > 20.f) ? x : log1pf(__expf(x));
}
__device__ __forceinline__ float act_gelu(float x) {
  return 0.5f * x * (1.f + erff(x * 0.70710678118654752f));
}

// ---------------- WMMA GEMM ------------------------------------------------
// C[M][N] = act(A[M][K](bf16, lda) * Wt^T + bias), C fp32 (ldc==N),
// optional bf16 mirror Cb. Wave tile: 32(M) x 32(N), 4 wmma per K-step.
// Requires: M%32==0, N%32==0, K%32==0 (caller pads). ACT: 0 none,1 softplus,2 gelu
template <int ACT>
__global__ void gemm_bf16_wmma(const u16* __restrict__ A, const u16* __restrict__ Wt,
                               const float* __restrict__ bias,
                               float* __restrict__ C, u16* __restrict__ Cb,
                               int M, int N, int K, int lda) {
  int lane = threadIdx.x & 31;
  int wave = threadIdx.x >> 5;
  int ntn  = N >> 5;
  int ntm  = M >> 5;
  int tile = blockIdx.x * 8 + wave;
  if (tile >= ntm * ntn) return;                 // wave-uniform: EXEC stays full
  int m0 = (tile / ntn) << 5;
  int n0 = (tile % ntn) << 5;

  v8f acc00 = {0.f,0.f,0.f,0.f,0.f,0.f,0.f,0.f};
  v8f acc01 = acc00, acc10 = acc00, acc11 = acc00;

  for (int k0 = 0; k0 < K; k0 += 32) {
    v16bf a0 = load_a_frag(A, lda, m0,      k0, lane);
    v16bf a1 = load_a_frag(A, lda, m0 + 16, k0, lane);
    v16bf b0 = load_b_frag(Wt, K, n0,      k0, lane);
    v16bf b1 = load_b_frag(Wt, K, n0 + 16, k0, lane);
    acc00 = __builtin_amdgcn_wmma_f32_16x16x32_bf16(false, a0, false, b0,
                                                    (short)0, acc00, false, false);
    acc01 = __builtin_amdgcn_wmma_f32_16x16x32_bf16(false, a0, false, b1,
                                                    (short)0, acc01, false, false);
    acc10 = __builtin_amdgcn_wmma_f32_16x16x32_bf16(false, a1, false, b0,
                                                    (short)0, acc10, false, false);
    acc11 = __builtin_amdgcn_wmma_f32_16x16x32_bf16(false, a1, false, b1,
                                                    (short)0, acc11, false, false);
  }

  int nn0 = n0 + (lane & 15);
  int nn1 = nn0 + 16;
  float bv0 = bias ? bias[nn0] : 0.f;
  float bv1 = bias ? bias[nn1] : 0.f;
  int mb = m0 + ((lane & 16) ? 8 : 0);
#pragma unroll
  for (int r = 0; r < 8; ++r) {
    float v00 = acc00[r] + bv0, v01 = acc01[r] + bv1;
    float v10 = acc10[r] + bv0, v11 = acc11[r] + bv1;
    if (ACT == 1) { v00 = act_softplus(v00); v01 = act_softplus(v01);
                    v10 = act_softplus(v10); v11 = act_softplus(v11); }
    if (ACT == 2) { v00 = act_gelu(v00); v01 = act_gelu(v01);
                    v10 = act_gelu(v10); v11 = act_gelu(v11); }
    size_t r0 = (size_t)(mb + r) * N;
    size_t r1 = (size_t)(mb + 16 + r) * N;
    C[r0 + nn0] = v00; C[r0 + nn1] = v01;
    C[r1 + nn0] = v10; C[r1 + nn1] = v11;
    if (Cb) {
      Cb[r0 + nn0] = f2bf(v00); Cb[r0 + nn1] = f2bf(v01);
      Cb[r1 + nn0] = f2bf(v10); Cb[r1 + nn1] = f2bf(v11);
    }
  }
}

// ---------------- Elementwise / small kernels ------------------------------

// W[K][N] fp32 -> Wt[Np][Kp] bf16, zero-padded outside K x N
__global__ void k_transpose_bf16(const float* __restrict__ W, u16* __restrict__ Wt,
                                 int K, int N, int Kp, int Np) {
  int i = blockIdx.x * 256 + threadIdx.x;
  if (i >= Kp * Np) return;
  int n = i / Kp, k = i % Kp;
  float v = (n < N && k < K) ? W[(size_t)k * N + n] : 0.f;
  Wt[(size_t)n * Kp + k] = f2bf(v);
}

// concat(img_id, img_attr) -> bf16 [NB,1536]
__global__ void k_cat(const float* __restrict__ a, const float* __restrict__ b,
                      u16* __restrict__ out) {
  int i = blockIdx.x * 256 + threadIdx.x;
  if (i >= NB * 2 * DIMX) return;
  int r = i / (2 * DIMX), j = i % (2 * DIMX);
  float v = (j < DIMX) ? a[(size_t)r * DIMX + j] : b[(size_t)r * DIMX + j - DIMX];
  out[i] = f2bf(v);
}

// conflict MLP: 1 -> 192 -> 384 -> 768 sigmoid; one block per batch row
__global__ void k_conflict(const float* __restrict__ cs,
                           const float* __restrict__ w1, const float* __restrict__ b1,
                           const float* __restrict__ w2, const float* __restrict__ b2,
                           const float* __restrict__ w3, const float* __restrict__ b3,
                           float* __restrict__ cw) {
  __shared__ float h1[192];
  __shared__ float h2[384];
  int b = blockIdx.x, tid = threadIdx.x;
  float c = cs[b];
  for (int j = tid; j < 192; j += 256) {
    float a = c * w1[j] + b1[j];
    h1[j] = a > 0.f ? a : 0.f;
  }
  __syncthreads();
  for (int j = tid; j < 384; j += 256) {
    float a = b2[j];
    for (int i = 0; i < 192; ++i) a += h1[i] * w2[i * 384 + j];
    h2[j] = a > 0.f ? a : 0.f;
  }
  __syncthreads();
  for (int j = tid; j < 768; j += 256) {
    float a = b3[j];
    for (int i = 0; i < 384; ++i) a += h2[i] * w3[i * 768 + j];
    cw[(size_t)b * 768 + j] = 1.f / (1.f + __expf(-a));
  }
}

// softmax gates over 2 modalities + conflict weighting + build s (fp32 & bf16)
__global__ void k_gate(const float* __restrict__ raw, const float* __restrict__ cw,
                       const float* __restrict__ img_id, const float* __restrict__ img_attr,
                       const float* __restrict__ txt_id, const float* __restrict__ txt_attr,
                       float* __restrict__ s, u16* __restrict__ s_bf) {
  int i = blockIdx.x * 256 + threadIdx.x;
  if (i >= NB * DIMX) return;
  int b = i / DIMX, j = i % DIMX;
  float r0 = raw[(size_t)b * 1536 + j];
  float r1 = raw[(size_t)b * 1536 + 768 + j];
  float mx = fmaxf(r0, r1);
  float e0 = __expf(r0 - mx), e1 = __expf(r1 - mx);
  float inv = 1.f / (e0 + e1);
  float c = cw[i];
  float v0 = img_id[i] * (e0 * inv) * c;
  float v1 = img_attr[i] * (e1 * inv) * c;
  float v2 = txt_id[i], v3 = txt_attr[i];
  size_t base = (size_t)b * 4 * DIMX + j;
  s[base] = v0; s[base + DIMX] = v1; s[base + 2 * DIMX] = v2; s[base + 3 * DIMX] = v3;
  s_bf[base] = f2bf(v0); s_bf[base + DIMX] = f2bf(v1);
  s_bf[base + 2 * DIMX] = f2bf(v2); s_bf[base + 3 * DIMX] = f2bf(v3);
}

// LayerNorm(768) with affine -> bf16; one block per row (RX rows)
__global__ void k_ln_bf16(const float* __restrict__ x, const float* __restrict__ g,
                          const float* __restrict__ bta, u16* __restrict__ out) {
  __shared__ float red[256];
  int row = blockIdx.x, tid = threadIdx.x;
  const float* xr = x + (size_t)row * DIMX;
  float v0 = xr[tid], v1 = xr[tid + 256], v2 = xr[tid + 512];
  red[tid] = v0 + v1 + v2;
  __syncthreads();
  for (int o = 128; o > 0; o >>= 1) { if (tid < o) red[tid] += red[tid + o]; __syncthreads(); }
  float mean = red[0] * (1.f / 768.f);
  __syncthreads();
  red[tid] = v0 * v0 + v1 * v1 + v2 * v2;
  __syncthreads();
  for (int o = 128; o > 0; o >>= 1) { if (tid < o) red[tid] += red[tid + o]; __syncthreads(); }
  float var  = red[0] * (1.f / 768.f) - mean * mean;
  float rstd = rsqrtf(var + 1e-5f);
  u16* orow = out + (size_t)row * DIMX;
  orow[tid]       = f2bf((v0 - mean) * rstd * g[tid]       + bta[tid]);
  orow[tid + 256] = f2bf((v1 - mean) * rstd * g[tid + 256] + bta[tid + 256]);
  orow[tid + 512] = f2bf((v2 - mean) * rstd * g[tid + 512] + bta[tid + 512]);
}

// causal depthwise conv (DC=4) over (direction-ordered) time + SiLU
__global__ void k_conv(const float* __restrict__ xz, const float* __restrict__ cwt,
                       const float* __restrict__ cb, float* __restrict__ xc,
                       u16* __restrict__ xcb, int dir) {
  int i = blockIdx.x * 256 + threadIdx.x;
  if (i >= RX * DIX) return;
  int d   = i % DIX;
  int tau = (i / DIX) & 3;
  int b   = i / (DIX * LX);
  float acc = cb[d];
#pragma unroll
  for (int k = 0; k < 4; ++k) {
    int idx = tau - 3 + k;
    if (idx >= 0) {
      int t = dir ? (3 - idx) : idx;
      acc += xz[((size_t)(b * 4 + t)) * 3072 + d] * cwt[d * 4 + k];
    }
  }
  float sv = acc / (1.f + __expf(-acc));   // silu
  xc[i]  = sv;
  xcb[i] = f2bf(sv);
}

// selective-scan over 4 steps, 16 states; fuse z-gating; output bf16 (GEMM input)
__global__ void k_scan(const float* __restrict__ dtb, const float* __restrict__ xc,
                       const float* __restrict__ proj, const float* __restrict__ xz,
                       const float* __restrict__ Alog, const float* __restrict__ Dres,
                       u16* __restrict__ yg, int dir) {
  int i = blockIdx.x * 256 + threadIdx.x;
  if (i >= NB * DIX) return;
  int d = i % DIX, b = i / DIX;
  float Av[16];
#pragma unroll
  for (int s = 0; s < 16; ++s) Av[s] = -__expf(Alog[d * 16 + s]);
  float Dv = Dres[d];
  float h[16];
#pragma unroll
  for (int s = 0; s < 16; ++s) h[s] = 0.f;
#pragma unroll
  for (int tau = 0; tau < 4; ++tau) {
    size_t row = (size_t)b * 4 + tau;
    float dtv = dtb[row * DIX + d];
    float xv  = xc[row * DIX + d];
    const float* pj = proj + row * PJLD;
    float dx = dtv * xv;
    float y  = 0.f;
#pragma unroll
    for (int s = 0; s < 16; ++s) {
      h[s] = __expf(dtv * Av[s]) * h[s] + dx * pj[48 + s];
      y += h[s] * pj[64 + s];
    }
    y += Dv * xv;
    int t0 = dir ? (3 - tau) : tau;
    float zv = xz[((size_t)b * 4 + t0) * 3072 + 1536 + d];
    float sz = zv / (1.f + __expf(-zv));
    yg[row * DIX + d] = f2bf(y * sz);
  }
}

// sf = s + s_fwd + s_bwd(time reversed) + inj; LayerNorm(no affine) -> bf16
__global__ void k_combine_ln(const float* __restrict__ s, const float* __restrict__ sf,
                             const float* __restrict__ sb, const float* __restrict__ sinj,
                             u16* __restrict__ out) {
  __shared__ float red[256];
  int row = blockIdx.x, tid = threadIdx.x;
  int b = row >> 2, t = row & 3;
  size_t rb = ((size_t)b * 4 + (3 - t)) * DIMX;
  size_t r0 = (size_t)row * DIMX;
  float v[3], ls = 0.f, lq = 0.f;
#pragma unroll
  for (int q = 0; q < 3; ++q) {
    int j = tid + q * 256;
    float x = s[r0 + j] + sf[r0 + j] + sb[rb + j] + sinj[r0 + j];
    v[q] = x; ls += x; lq += x * x;
  }
  red[tid] = ls;
  __syncthreads();
  for (int o = 128; o > 0; o >>= 1) { if (tid < o) red[tid] += red[tid + o]; __syncthreads(); }
  float mean = red[0] * (1.f / 768.f);
  __syncthreads();
  red[tid] = lq;
  __syncthreads();
  for (int o = 128; o > 0; o >>= 1) { if (tid < o) red[tid] += red[tid + o]; __syncthreads(); }
  float var  = red[0] * (1.f / 768.f) - mean * mean;
  float rstd = rsqrtf(var + 1e-5f);
  u16* orow = out + r0;
#pragma unroll
  for (int q = 0; q < 3; ++q) orow[tid + q * 256] = f2bf((v[q] - mean) * rstd);
}

// mean over 4 time steps -> bf16
__global__ void k_pool(const float* __restrict__ so, u16* __restrict__ pooled) {
  int i = blockIdx.x * 256 + threadIdx.x;
  if (i >= NB * DIMX) return;
  int b = i / DIMX, j = i % DIMX;
  size_t base = (size_t)b * 4 * DIMX + j;
  float m = 0.25f * (so[base] + so[base + DIMX] + so[base + 2 * DIMX] + so[base + 3 * DIMX]);
  pooled[i] = f2bf(m);
}

// final LayerNorm over 256 with affine -> d_out (fp32)
__global__ void k_final_ln(const float* __restrict__ x, const float* __restrict__ g,
                           const float* __restrict__ bta, float* __restrict__ out) {
  __shared__ float red[256];
  int b = blockIdx.x, tid = threadIdx.x;
  float v = x[(size_t)b * 256 + tid];
  red[tid] = v;
  __syncthreads();
  for (int o = 128; o > 0; o >>= 1) { if (tid < o) red[tid] += red[tid + o]; __syncthreads(); }
  float mean = red[0] * (1.f / 256.f);
  __syncthreads();
  red[tid] = v * v;
  __syncthreads();
  for (int o = 128; o > 0; o >>= 1) { if (tid < o) red[tid] += red[tid + o]; __syncthreads(); }
  float var  = red[0] * (1.f / 256.f) - mean * mean;
  float rstd = rsqrtf(var + 1e-5f);
  out[(size_t)b * 256 + tid] = (v - mean) * rstd * g[tid] + bta[tid];
}

// ---------------- host launcher --------------------------------------------

static void run_gemm(int act, hipStream_t stream,
                     const u16* A, const u16* Wt, const float* bias,
                     float* C, u16* Cb, int M, int N, int K, int lda) {
  int tiles  = (M >> 5) * (N >> 5);
  int blocks = (tiles + 7) >> 3;
  if (act == 1)      gemm_bf16_wmma<1><<<blocks, 256, 0, stream>>>(A, Wt, bias, C, Cb, M, N, K, lda);
  else if (act == 2) gemm_bf16_wmma<2><<<blocks, 256, 0, stream>>>(A, Wt, bias, C, Cb, M, N, K, lda);
  else               gemm_bf16_wmma<0><<<blocks, 256, 0, stream>>>(A, Wt, bias, C, Cb, M, N, K, lda);
}

extern "C" void kernel_launch(void* const* d_in, const int* in_sizes, int n_in,
                              void* d_out, int out_size, void* d_ws, size_t ws_size,
                              hipStream_t stream) {
  (void)in_sizes; (void)n_in; (void)out_size; (void)ws_size;

  const float* img_id   = (const float*)d_in[0];
  const float* img_attr = (const float*)d_in[1];
  const float* txt_id   = (const float*)d_in[2];
  const float* txt_attr = (const float*)d_in[3];
  const float* cscore   = (const float*)d_in[4];
  const float* cm1_w = (const float*)d_in[5];  const float* cm1_b = (const float*)d_in[6];
  const float* cm2_w = (const float*)d_in[7];  const float* cm2_b = (const float*)d_in[8];
  const float* cm3_w = (const float*)d_in[9];  const float* cm3_b = (const float*)d_in[10];
  const float* mp_w  = (const float*)d_in[11]; const float* mp_b  = (const float*)d_in[12];
  const float* ln_g  = (const float*)d_in[13]; const float* ln_b  = (const float*)d_in[14];
  const float* inj_w = (const float*)d_in[15]; const float* inj_b = (const float*)d_in[16];
  const float* bn1_w = (const float*)d_in[17]; const float* bn1_b = (const float*)d_in[18];
  const float* bn2_w = (const float*)d_in[19]; const float* bn2_b = (const float*)d_in[20];
  const float* op_w  = (const float*)d_in[21]; const float* op_b  = (const float*)d_in[22];
  const float* on_g  = (const float*)d_in[23]; const float* on_b  = (const float*)d_in[24];
  // per-direction params: f_ at 25.., b_ at 34..
  const float* in_w_[2]   = {(const float*)d_in[25], (const float*)d_in[34]};
  const float* conv_w_[2] = {(const float*)d_in[26], (const float*)d_in[35]};
  const float* conv_b_[2] = {(const float*)d_in[27], (const float*)d_in[36]};
  const float* xp_w_[2]   = {(const float*)d_in[28], (const float*)d_in[37]};
  const float* dt_w_[2]   = {(const float*)d_in[29], (const float*)d_in[38]};
  const float* dt_b_[2]   = {(const float*)d_in[30], (const float*)d_in[39]};
  const float* Alog_[2]   = {(const float*)d_in[31], (const float*)d_in[40]};
  const float* Dres_[2]   = {(const float*)d_in[32], (const float*)d_in[41]};
  const float* out_w_[2]  = {(const float*)d_in[33], (const float*)d_in[42]};

  char* ws = (char*)d_ws;
  size_t off = 0;
  auto alloc = [&](size_t bytes) -> char* {
    char* p = ws + off;
    off += (bytes + 255) & ~(size_t)255;
    return p;
  };

  // bf16 transposed (zero-padded) weights
  u16* w_mpT   = (u16*)alloc((size_t)1536 * 1536 * 2);
  u16* w_injT  = (u16*)alloc((size_t)768 * 768 * 2);
  u16* w_bn1T  = (u16*)alloc((size_t)3072 * 768 * 2);
  u16* w_bn2T  = (u16*)alloc((size_t)768 * 3072 * 2);
  u16* w_opT   = (u16*)alloc((size_t)256 * 768 * 2);
  u16* w_inT[2], *w_xpT[2], *w_dtT[2], *w_outT[2];
  for (int d = 0; d < 2; ++d) {
    w_inT[d]  = (u16*)alloc((size_t)3072 * 768 * 2);
    w_xpT[d]  = (u16*)alloc((size_t)PJLD * 1536 * 2);   // N 80->96 padded
    w_dtT[d]  = (u16*)alloc((size_t)1536 * 64 * 2);     // K 48->64 padded
    w_outT[d] = (u16*)alloc((size_t)768 * 1536 * 2);
  }
  // bf16 activations
  u16* a_cat     = (u16*)alloc((size_t)NB * 1536 * 2);
  u16* s_bf      = (u16*)alloc((size_t)RX * DIMX * 2);
  u16* xn_bf     = (u16*)alloc((size_t)RX * DIMX * 2);
  u16* xc_bf     = (u16*)alloc((size_t)RX * DIX * 2);
  u16* proj_bf   = (u16*)alloc((size_t)RX * PJLD * 2);
  u16* yg_bf     = (u16*)alloc((size_t)RX * DIX * 2);
  u16* sfn_bf    = (u16*)alloc((size_t)RX * DIMX * 2);
  u16* g1_bf     = (u16*)alloc((size_t)RX * 3072 * 2);
  u16* pooled_bf = (u16*)alloc((size_t)NB * DIMX * 2);
  // fp32 activations
  float* raw    = (float*)alloc((size_t)NB * 1536 * 4);
  float* s      = (float*)alloc((size_t)RX * DIMX * 4);
  float* xz     = (float*)alloc((size_t)RX * 3072 * 4);   // reused as g1 later
  float* xc     = (float*)alloc((size_t)RX * DIX * 4);
  float* proj   = (float*)alloc((size_t)RX * PJLD * 4);
  float* dtb    = (float*)alloc((size_t)RX * DIX * 4);
  float* s_dir0 = (float*)alloc((size_t)RX * DIMX * 4);
  float* s_dir1 = (float*)alloc((size_t)RX * DIMX * 4);
  float* sinj   = (float*)alloc((size_t)RX * DIMX * 4);
  float* s_out  = (float*)alloc((size_t)RX * DIMX * 4);
  float* outf   = (float*)alloc((size_t)NB * 256 * 4);
  float* g1     = xz;                                     // reuse (xz dead by bn1)
  float* s_dir[2] = {s_dir0, s_dir1};

  float* out_ln = (float*)d_out;                     // [NB,256]
  float* out_cw = (float*)d_out + (size_t)NB * 256;  // [NB,768]

  // ---- weight conversion (transpose to N-major bf16, padded) ----
  k_transpose_bf16<<<CDIV(1536 * 1536, 256), 256, 0, stream>>>(mp_w,  w_mpT,  1536, 1536, 1536, 1536);
  k_transpose_bf16<<<CDIV(768 * 768,   256), 256, 0, stream>>>(inj_w, w_injT, 768,  768,  768,  768);
  k_transpose_bf16<<<CDIV(768 * 3072,  256), 256, 0, stream>>>(bn1_w, w_bn1T, 768,  3072, 768,  3072);
  k_transpose_bf16<<<CDIV(3072 * 768,  256), 256, 0, stream>>>(bn2_w, w_bn2T, 3072, 768,  3072, 768);
  k_transpose_bf16<<<CDIV(768 * 256,   256), 256, 0, stream>>>(op_w,  w_opT,  768,  256,  768,  256);
  for (int d = 0; d < 2; ++d) {
    k_transpose_bf16<<<CDIV(768 * 3072,   256), 256, 0, stream>>>(in_w_[d],  w_inT[d],  768,  3072, 768,  3072);
    k_transpose_bf16<<<CDIV(1536 * PJLD,  256), 256, 0, stream>>>(xp_w_[d],  w_xpT[d],  1536, 80,   1536, PJLD);
    k_transpose_bf16<<<CDIV(64 * 1536,    256), 256, 0, stream>>>(dt_w_[d],  w_dtT[d],  48,   1536, 64,   1536);
    k_transpose_bf16<<<CDIV(1536 * 768,   256), 256, 0, stream>>>(out_w_[d], w_outT[d], 1536, 768,  1536, 768);
  }

  // ---- front end ----
  k_cat<<<CDIV(NB * 1536, 256), 256, 0, stream>>>(img_id, img_attr, a_cat);
  k_conflict<<<NB, 256, 0, stream>>>(cscore, cm1_w, cm1_b, cm2_w, cm2_b, cm3_w, cm3_b, out_cw);
  run_gemm(0, stream, a_cat, w_mpT, mp_b, raw, nullptr, NB, 1536, 1536, 1536);
  k_gate<<<CDIV(NB * DIMX, 256), 256, 0, stream>>>(raw, out_cw, img_id, img_attr,
                                                   txt_id, txt_attr, s, s_bf);
  k_ln_bf16<<<RX, 256, 0, stream>>>(s, ln_g, ln_b, xn_bf);

  // ---- bidirectional mamba ----
  for (int d = 0; d < 2; ++d) {
    run_gemm(0, stream, xn_bf, w_inT[d], nullptr, xz, nullptr, RX, 3072, 768, 768);
    k_conv<<<CDIV(RX * DIX, 256), 256, 0, stream>>>(xz, conv_w_[d], conv_b_[d], xc, xc_bf, d);
    // proj: N padded 80->96 (extra cols are zero-weight), K=1536
    run_gemm(0, stream, xc_bf, w_xpT[d], nullptr, proj, proj_bf, RX, PJLD, 1536, 1536);
    // dt: K padded 48->64 (A cols 48..63 hit zero weights), lda = PJLD
    run_gemm(1, stream, proj_bf, w_dtT[d], dt_b_[d], dtb, nullptr, RX, 1536, 64, PJLD);
    k_scan<<<CDIV(NB * DIX, 256), 256, 0, stream>>>(dtb, xc, proj, xz,
                                                    Alog_[d], Dres_[d], yg_bf, d);
    run_gemm(0, stream, yg_bf, w_outT[d], nullptr, s_dir[d], nullptr, RX, 768, 1536, 1536);
  }

  // ---- fusion + MLP head ----
  run_gemm(0, stream, s_bf, w_injT, inj_b, sinj, nullptr, RX, 768, 768, 768);
  k_combine_ln<<<RX, 256, 0, stream>>>(s, s_dir[0], s_dir[1], sinj, sfn_bf);
  run_gemm(2, stream, sfn_bf, w_bn1T, bn1_b, g1, g1_bf, RX, 3072, 768, 768);
  run_gemm(0, stream, g1_bf, w_bn2T, bn2_b, s_out, nullptr, RX, 768, 3072, 3072);
  k_pool<<<CDIV(NB * DIMX, 256), 256, 0, stream>>>(s_out, pooled_bf);
  run_gemm(0, stream, pooled_bf, w_opT, op_b, outf, nullptr, NB, 256, 768, 768);
  k_final_ln<<<NB, 256, 0, stream>>>(outf, on_g, on_b, out_ln);
}
